// BertForSpanAspectExtraction_72971494359130
// MI455X (gfx1250) — matrix operand order repaired and checked by reading, not compiled, not run on hardware
//
#include <hip/hip_runtime.h>
#include <hip/hip_bf16.h>
#include <math.h>
#include <stdint.h>

#define B_  4
#define L_  256
#define H_  768
#define H2_ 384
#define M_  (B_*L_)   // 1024 token rows

typedef __attribute__((ext_vector_type(16))) _Float16 v16h;
typedef __attribute__((ext_vector_type(8)))  float    v8f;

#define LDS_AS __attribute__((address_space(3)))
#define GBL_AS __attribute__((address_space(1)))

// ---------------------------------------------------------------------------
// Kernel 1: start/end logits  (two mat-vecs over H, one block per token row)
// ---------------------------------------------------------------------------
__global__ void logits_kernel(const float* __restrict__ x,
                              const float* __restrict__ w_start, const float* __restrict__ b_start,
                              const float* __restrict__ w_end,   const float* __restrict__ b_end,
                              float* __restrict__ out)
{
    __shared__ float red_s[256];
    __shared__ float red_e[256];
    int row = blockIdx.x;          // b*L + l  in [0, 1024)
    int t   = threadIdx.x;
    const float* xr = x + (size_t)row * H_;
    float ss = 0.f, se = 0.f;
    #pragma unroll
    for (int c = 0; c < H_ / 256; ++c) {
        float v = xr[t + c * 256];
        ss = fmaf(v, w_start[t + c * 256], ss);
        se = fmaf(v, w_end  [t + c * 256], se);
    }
    red_s[t] = ss; red_e[t] = se;
    __syncthreads();
    for (int off = 128; off > 0; off >>= 1) {
        if (t < off) { red_s[t] += red_s[t + off]; red_e[t] += red_e[t + off]; }
        __syncthreads();
    }
    if (t == 0) {
        out[row]      = red_s[0] + b_start[0];
        out[M_ + row] = red_e[0] + b_end[0];
    }
}

// ---------------------------------------------------------------------------
// Kernel 2: s_proj = x @ W1s, e_proj = x @ W1e via V_WMMA_F32_16X16X32_F16.
// Block = 8 waves = 128 M-rows x one 64-col N-strip.
//   * W tile (32x64 f32) staged into LDS per K-step via the async-to-LDS
//     engine (ASYNCcnt) when available, shared by all 8 waves.
//   * Each wave: 1 A fragment (two float4 pairs) feeds 4 WMMAs per K-step.
//   * LDS stride 65 -> bank-conflict-free column reads for the B fragment.
// ---------------------------------------------------------------------------
#define WS_ 65   // padded LDS row stride for the 32x64 W tile

__global__ void proj_wmma_kernel(const float* __restrict__ x,
                                 const float* __restrict__ W1s,
                                 const float* __restrict__ W1e,
                                 float* __restrict__ sproj,
                                 float* __restrict__ eproj)
{
    __shared__ float wtile[32 * WS_];

    int strip = blockIdx.x >> 3;        // 0..11 (6 strips W1s, 6 strips W1e)
    int mblk  = blockIdx.x & 7;         // 0..7
    int w     = threadIdx.x >> 5;       // wave id in block
    int lane  = threadIdx.x & 31;
    int half  = lane >> 4;              // 0: lanes 0-15, 1: lanes 16-31
    int l15   = lane & 15;

    const float* W    = (strip < 6) ? W1s   : W1e;
    float*       outp = (strip < 6) ? sproj : eproj;
    int n0 = ((strip < 6) ? strip : strip - 6) * 64;   // col base in [0,384)
    int m0 = (mblk * 8 + w) * 16;                      // row base in [0,1024)
    int mr = m0 + l15;                                 // A row owned by lane

    v8f acc[4] = {{}, {}, {}, {}};

    for (int kk = 0; kk < H_ / 32; ++kk) {
        int k0 = kk * 32;

        // ---- stage W[k0:k0+32, n0:n0+64] into LDS (coalesced) ----
        #pragma unroll
        for (int idx = 0; idx < 8; ++idx) {
            int flat = threadIdx.x + idx * 256;        // 0..2047
            int kr = flat >> 6, cn = flat & 63;
            const float* gp = &W[(size_t)(k0 + kr) * H2_ + n0 + cn];
            float*       lp = &wtile[kr * WS_ + cn];
#if __has_builtin(__builtin_amdgcn_global_load_async_to_lds_b32)
            // gfx1250 async-to-LDS path (tracked by ASYNCcnt)
            __builtin_amdgcn_global_load_async_to_lds_b32(
                (GBL_AS int*)(uintptr_t)gp,
                (LDS_AS int*)(unsigned)(uintptr_t)lp,
                0, 0);
#else
            *lp = *gp;
#endif
        }
#if __has_builtin(__builtin_amdgcn_global_load_async_to_lds_b32)
#if __has_builtin(__builtin_amdgcn_s_wait_asynccnt)
        __builtin_amdgcn_s_wait_asynccnt(0);
#else
        asm volatile("s_wait_asynccnt 0x0" ::: "memory");
#endif
#endif
        __syncthreads();

        // ---- A fragment: lane<16 -> K {0..7,16..23}; lane>=16 -> {8..15,24..31}
        const float* xa = x + (size_t)mr * H_ + k0 + half * 8;
        float4 a0 = *(const float4*)(xa);
        float4 a1 = *(const float4*)(xa + 4);
        float4 a2 = *(const float4*)(xa + 16);
        float4 a3 = *(const float4*)(xa + 20);
        if (kk + 1 < H_ / 32)
            __builtin_prefetch(xa + 32, 0, 1);         // global_prefetch for next K-step
        v16h a;
        a[0]  = (_Float16)a0.x; a[1]  = (_Float16)a0.y;
        a[2]  = (_Float16)a0.z; a[3]  = (_Float16)a0.w;
        a[4]  = (_Float16)a1.x; a[5]  = (_Float16)a1.y;
        a[6]  = (_Float16)a1.z; a[7]  = (_Float16)a1.w;
        a[8]  = (_Float16)a2.x; a[9]  = (_Float16)a2.y;
        a[10] = (_Float16)a2.z; a[11] = (_Float16)a2.w;
        a[12] = (_Float16)a3.x; a[13] = (_Float16)a3.y;
        a[14] = (_Float16)a3.z; a[15] = (_Float16)a3.w;

        // ---- 4 WMMAs reuse the A fragment across the 64-col strip ----
        #pragma unroll
        for (int n = 0; n < 4; ++n) {
            // B fragment: lane<16 -> K {0..7,16..23}; lane>=16 -> K {8..15,24..31} (col n*16+l15)
            const float* bp = &wtile[(half * 8) * WS_ + n * 16 + l15];
            v16h bf;
            #pragma unroll
            for (int e = 0; e < 16; ++e)
                bf[e] = (_Float16)bp[(e + (e >= 8 ? 8 : 0)) * WS_];
            acc[n] = __builtin_amdgcn_wmma_f32_16x16x32_f16(
                         false, a, false, bf, (short)0, acc[n], false, false);
        }
        __syncthreads();   // before next K-step overwrites the LDS tile
    }

    // D 16x16 f32 layout: lane<16 -> (M=m0+v, N=lane); lane>=16 -> (M=m0+8+v)
    #pragma unroll
    for (int n = 0; n < 4; ++n) {
        float* orow = outp + (size_t)(m0 + half * 8) * H2_ + n0 + n * 16 + l15;
        #pragma unroll
        for (int v = 0; v < 8; ++v)
            orow[(size_t)v * H2_] = acc[n][v];
    }
}

// ---------------------------------------------------------------------------
// Kernel 3: span logits.  Block = 256 threads handles one (b, 16x16 ij-tile).
// span[b,i,j] = sigmoid( sum_k relu(s[i,k] + e[j,k] + b1[k]) * W2[k] + b2 )
// ---------------------------------------------------------------------------
#define SP 388   // padded LDS row stride (388 % 64 == 4 -> conflict-free)

__global__ void span_kernel(const float* __restrict__ sproj,
                            const float* __restrict__ eproj,
                            const float* __restrict__ b1,
                            const float* __restrict__ W2,
                            const float* __restrict__ b2,
                            float* __restrict__ out)
{
    __shared__ float s_lds[16 * SP];
    __shared__ float e_lds[16 * SP];
    __shared__ float w2s[H2_];
    int jt = blockIdx.x, it = blockIdx.y, b = blockIdx.z;
    int t  = threadIdx.x;

    // stage 16 s-rows (i-tile) and 16 (e+b1)-rows (j-tile)
    for (int idx = t; idx < 16 * H2_; idx += 256) {
        int r = idx / H2_, k = idx - r * H2_;
        s_lds[r * SP + k] = sproj[((size_t)(b * L_ + it * 16 + r)) * H2_ + k];
        e_lds[r * SP + k] = eproj[((size_t)(b * L_ + jt * 16 + r)) * H2_ + k] + b1[k];
    }
    for (int k = t; k < H2_; k += 256) w2s[k] = W2[k];
    __syncthreads();

    int iloc = t >> 4, jloc = t & 15;
    const float* srow = &s_lds[iloc * SP];
    const float* erow = &e_lds[jloc * SP];
    float acc = 0.f;
    #pragma unroll 8
    for (int k = 0; k < H2_; ++k) {
        float h = srow[k] + erow[k];
        h = fmaxf(h, 0.f);
        acc = fmaf(h, w2s[k], acc);
    }
    float z   = acc + b2[0];
    float sig = 1.f / (1.f + __expf(-z));
    int i = it * 16 + iloc, j = jt * 16 + jloc;
    out[((size_t)b * L_ + i) * L_ + j] = sig;
}

// ---------------------------------------------------------------------------
extern "C" void kernel_launch(void* const* d_in, const int* in_sizes, int n_in,
                              void* d_out, int out_size, void* d_ws, size_t ws_size,
                              hipStream_t stream)
{
    const float* x   = (const float*)d_in[0];
    const float* wst = (const float*)d_in[1];
    const float* bst = (const float*)d_in[2];
    const float* wen = (const float*)d_in[3];
    const float* ben = (const float*)d_in[4];
    const float* W1s = (const float*)d_in[5];
    const float* W1e = (const float*)d_in[6];
    const float* b1  = (const float*)d_in[7];
    const float* W2  = (const float*)d_in[8];
    const float* b2  = (const float*)d_in[9];
    float* out = (float*)d_out;

    // workspace: s_proj (1024x384 f32) followed by e_proj (1024x384 f32) = 3 MB
    float* sproj = (float*)d_ws;
    float* eproj = sproj + (size_t)M_ * H2_;

    logits_kernel<<<M_, 256, 0, stream>>>(x, wst, bst, wen, ben, out);

    // 12 N-strips * 8 M-blocks = 96 blocks of 8 waves
    proj_wmma_kernel<<<96, 256, 0, stream>>>(x, W1s, W1e, sproj, eproj);

    span_kernel<<<dim3(L_ / 16, L_ / 16, B_), 256, 0, stream>>>(
        sproj, eproj, b1, W2, b2, out + 2 * M_);
}
